// BiAttention_18056042512950
// MI455X (gfx1250) — matrix operand order repaired and checked by reading, not compiled, not run on hardware
//
#include <hip/hip_runtime.h>
#include <hip/hip_bf16.h>
#include <math.h>

// ---------------------------------------------------------------------------
// BiAttention forward for MI455X (gfx1250, wave32, WMMA).
// All heavy math runs through v_wmma_f32_16x16x32_bf16.
// GEMM: 256x128 block tile, 8 waves (4m x 2n), 64x64 wave tile (16 accums),
// double-buffered LDS, b128 staging, one barrier per 32-deep k-step.
// ---------------------------------------------------------------------------

typedef __attribute__((ext_vector_type(16))) __bf16 v16bf;
typedef __attribute__((ext_vector_type(8)))  float  v8f;

#define BDIM 48
#define SDIM 512
#define HDIM 768
#define NEGV (-1e9f)

// LDS tile row pitch: 32 bf16 payload + 8 pad = 40 bf16 (80 B, 16B aligned)
#define APITCH 40
#define APITCHD 20

static __device__ __forceinline__ unsigned short f2bf_rne(float f) {
    unsigned int u = __float_as_uint(f);
    unsigned int r = u + 0x7fffu + ((u >> 16) & 1u);
    return (unsigned short)(r >> 16);
}

// float -> bf16 conversion (grid-stride)
__global__ void f2bf_kernel(const float* __restrict__ in,
                            unsigned short* __restrict__ out, int n) {
    int i = blockIdx.x * blockDim.x + threadIdx.x;
    int stride = gridDim.x * blockDim.x;
    for (; i < n; i += stride) out[i] = f2bf_rne(in[i]);
}

// ---------------------------------------------------------------------------
// Fragment gathers from LDS (row pitch = APITCHD dwords).
// A fragment (16x32, MxK): lane L holds row M=L%16, half=L/16.
//   element e -> k = 8*half + (e&7) + (e>=8 ? 16 : 0)
// B fragment (32x16, KxN): lane L holds col N=L%16, half=L/16.
//   element e -> k = 8*half + (e&7) + (e>=8 ? 16 : 0)
// ---------------------------------------------------------------------------
static __device__ __forceinline__ v16bf load_afrag(const unsigned int* sAd,
                                                   int row, int hf) {
    union { unsigned int u[8]; v16bf v; } f;
    const unsigned int* p = sAd + row * APITCHD + hf * 4;
#pragma unroll
    for (int j = 0; j < 4; ++j) f.u[j] = p[j];         // k = 8*hf + 0..7
#pragma unroll
    for (int j = 0; j < 4; ++j) f.u[4 + j] = p[8 + j]; // k = 16 + 8*hf + 0..7
    return f.v;
}

static __device__ __forceinline__ v16bf load_bfrag(const unsigned int* sBd,
                                                   int nrow, int hf) {
    union { unsigned int u[8]; v16bf v; } f;
    const unsigned int* p = sBd + nrow * APITCHD + hf * 4;  // k = 8*hf + 0..7, 16 + 8*hf + 0..7
#pragma unroll
    for (int j = 0; j < 4; ++j) { f.u[j] = p[j]; f.u[4 + j] = p[8 + j]; }
    return f.v;
}

// ---------------------------------------------------------------------------
// Generic bf16 WMMA GEMM:  C = epi( scale * A(bf16[M,K]) * B(bf16[N,K])^T + bias )
// Block tile 256(M) x 128(N), K-step 32, 256 threads = 8 waves (4m x 2n),
// wave tile 64x64 = sixteen 16x16 f32 accumulators. Double-buffered LDS.
// OUTMODE: 0 = bf16 out, 1 = f32 out
// EPI:     0 = none, 1 = tanh, 2 = +resid (f32)
// TRANSOUT: write C transposed (bf16 only): C[n*ldc + m]
// ---------------------------------------------------------------------------
template <int OUTMODE, int EPI, bool TRANSOUT>
__global__ __launch_bounds__(256)
void gemm_bf16_wmma(const unsigned short* __restrict__ A, long long strideAb, int lda,
                    const unsigned short* __restrict__ Bw, long long strideBb, int ldb,
                    void* __restrict__ Cv, long long strideCb, int ldc,
                    const float* __restrict__ bias,
                    const float* __restrict__ resid, int ldr,
                    int Ktot, float scale) {
    __shared__ unsigned short sA[2][256 * APITCH];
    __shared__ unsigned short sB[2][128 * APITCH];

    const int tid    = threadIdx.x;
    const int lane   = tid & 31;
    const int wave   = tid >> 5;
    const int wave_m = wave >> 1;     // 0..3 -> 64 rows each
    const int wave_n = wave & 1;      // 0..1 -> 64 cols each
    const int l16    = lane & 15;
    const int hf     = lane >> 4;

    const int m0 = blockIdx.x * 256;
    const int n0 = blockIdx.y * 128;

    const unsigned short* Ab = A + (long long)blockIdx.z * strideAb;
    const unsigned short* Bb = Bw + (long long)blockIdx.z * strideBb;

    // staging: tile rows x 16 bf16; each row = 2 x uint4 (8 bf16 each)
    const int arow0 = tid >> 1;           // A rows: tid/2 and tid/2+128
    const int acol  = (tid & 1) * 8;      // bf16 column offset (uint4 granules)
    const int brow  = tid >> 1;           // B rows: tid/2
    v8f acc[4][4];
#pragma unroll
    for (int mi = 0; mi < 4; ++mi)
#pragma unroll
        for (int ni = 0; ni < 4; ++ni)
#pragma unroll
            for (int r = 0; r < 8; ++r) acc[mi][ni][r] = 0.0f;

    const int ksteps = Ktot >> 5;

    // ---- prologue: stage k-step 0 into buffer 0 ----
    {
        uint4 a0 = *(const uint4*)(Ab + (size_t)(m0 + arow0) * lda + acol);
        uint4 a1 = *(const uint4*)(Ab + (size_t)(m0 + arow0 + 128) * lda + acol);
        uint4 b0 = *(const uint4*)(Bb + (size_t)(n0 + brow) * ldb + acol);
        *(uint4*)(sA[0] + arow0 * APITCH + acol) = a0;
        *(uint4*)(sA[0] + (arow0 + 128) * APITCH + acol) = a1;
        *(uint4*)(sB[0] + brow * APITCH + acol) = b0;
    }
    __syncthreads();

    int cur = 0;
    for (int kt = 0; kt < ksteps; ++kt) {
        const bool pf = (kt + 1 < ksteps);
        uint4 a0, a1, b0;
        if (pf) {
            const int k1 = (kt + 1) << 5;
            a0 = *(const uint4*)(Ab + (size_t)(m0 + arow0) * lda + k1 + acol);
            a1 = *(const uint4*)(Ab + (size_t)(m0 + arow0 + 128) * lda + k1 + acol);
            b0 = *(const uint4*)(Bb + (size_t)(n0 + brow) * ldb + k1 + acol);
        }

        const unsigned int* sAd = (const unsigned int*)sA[cur];
        const unsigned int* sBd = (const unsigned int*)sB[cur];
        v16bf bfr[4];
#pragma unroll
        for (int ni = 0; ni < 4; ++ni)
            bfr[ni] = load_bfrag(sBd, wave_n * 64 + ni * 16 + l16, hf);
#pragma unroll
        for (int mi = 0; mi < 4; ++mi) {
            v16bf af = load_afrag(sAd, wave_m * 64 + mi * 16 + l16, hf);
#pragma unroll
            for (int ni = 0; ni < 4; ++ni) {
                acc[mi][ni] = __builtin_amdgcn_wmma_f32_16x16x32_bf16(
                    false, af, false, bfr[ni], (short)0, acc[mi][ni], false, false);
            }
        }

        if (pf) {
            unsigned short* nA = sA[cur ^ 1];
            unsigned short* nB = sB[cur ^ 1];
            *(uint4*)(nA + arow0 * APITCH + acol) = a0;
            *(uint4*)(nA + (arow0 + 128) * APITCH + acol) = a1;
            *(uint4*)(nB + brow * APITCH + acol) = b0;
            __syncthreads();
            cur ^= 1;
        }
    }

    // -------- epilogue --------
#pragma unroll
    for (int mi = 0; mi < 4; ++mi) {
#pragma unroll
        for (int ni = 0; ni < 4; ++ni) {
            const int ng = n0 + wave_n * 64 + ni * 16 + l16;
            float bval = bias ? bias[ng] : 0.0f;
            const int mbase = m0 + wave_m * 64 + mi * 16 + hf * 8;
            if (TRANSOUT) {
                union { unsigned short s[8]; uint4 q; } o;
#pragma unroll
                for (int r = 0; r < 8; ++r) {
                    float v = acc[mi][ni][r] * scale + bval;
                    if (EPI == 1) v = tanhf(v);
                    o.s[r] = f2bf_rne(v);
                }
                unsigned short* Ct = (unsigned short*)Cv + (long long)blockIdx.z * strideCb;
                *(uint4*)(Ct + (size_t)ng * ldc + mbase) = o.q;
            } else {
#pragma unroll
                for (int r = 0; r < 8; ++r) {
                    float v = acc[mi][ni][r] * scale + bval;
                    if (EPI == 1) v = tanhf(v);
                    const int mg = mbase + r;
                    if (EPI == 2) v += resid[(size_t)mg * ldr + ng];
                    if (OUTMODE == 0) {
                        unsigned short* Cb =
                            (unsigned short*)Cv + (long long)blockIdx.z * strideCb;
                        Cb[(size_t)mg * ldc + ng] = f2bf_rne(v);
                    } else {
                        float* Cf = (float*)Cv + (long long)blockIdx.z * strideCb;
                        Cf[(size_t)mg * ldc + ng] = v;
                    }
                }
            }
        }
    }
}

// ---------------------------------------------------------------------------
// Masked softmax over scores rows. One wave per row (512 keys, 16/lane).
// scores already scaled by 1/sqrt(H). Applies cross-segment allow-mask and
// additive pad mask on q->context edges, softmax, writes bf16 probabilities.
// ---------------------------------------------------------------------------
__global__ __launch_bounds__(256)
void softmax_mask_kernel(const float* __restrict__ Sc,
                         unsigned short* __restrict__ P,
                         const float* __restrict__ mask,
                         const int* __restrict__ divide_pos) {
    const int wave = threadIdx.x >> 5;
    const int lane = threadIdx.x & 31;
    const int row  = blockIdx.x * 8 + wave;       // 0..B*S-1
    const int b    = row >> 9;
    const int q    = row & 511;
    const int d    = divide_pos[b];
    const bool isq = (q < d);

    const float* src  = Sc + (size_t)row * SDIM + lane * 16;
    const float* mrow = mask + (size_t)b * SDIM;

    float v[16];
#pragma unroll
    for (int i = 0; i < 4; ++i) {
        float4 f = ((const float4*)src)[i];
        v[4 * i + 0] = f.x; v[4 * i + 1] = f.y;
        v[4 * i + 2] = f.z; v[4 * i + 3] = f.w;
    }

    float mx = -3.4e38f;
#pragma unroll
    for (int j = 0; j < 16; ++j) {
        const int k = lane * 16 + j;
        const bool isk = (k < d);
        float add = (isq && !isk) ? mrow[k] : 0.0f;
        float val = (isq != isk) ? (v[j] + add) : NEGV;
        v[j] = val;
        mx = fmaxf(mx, val);
    }
#pragma unroll
    for (int off = 16; off >= 1; off >>= 1)
        mx = fmaxf(mx, __shfl_xor(mx, off, 32));

    float s = 0.0f;
#pragma unroll
    for (int j = 0; j < 16; ++j) {
        float e = __expf(v[j] - mx);
        v[j] = e;
        s += e;
    }
#pragma unroll
    for (int off = 16; off >= 1; off >>= 1)
        s += __shfl_xor(s, off, 32);
    const float inv = 1.0f / s;

    unsigned short* dst = P + (size_t)row * SDIM + lane * 16;
#pragma unroll
    for (int hblk = 0; hblk < 2; ++hblk) {
        union { unsigned short h[8]; uint4 q4; } o;
#pragma unroll
        for (int j = 0; j < 8; ++j) o.h[j] = f2bf_rne(v[hblk * 8 + j] * inv);
        ((uint4*)dst)[hblk] = o.q4;
    }
}

// ---------------------------------------------------------------------------
extern "C" void kernel_launch(void* const* d_in, const int* in_sizes, int n_in,
                              void* d_out, int out_size, void* d_ws, size_t ws_size,
                              hipStream_t stream) {
    (void)in_sizes; (void)n_in; (void)out_size; (void)ws_size;
    const float* x    = (const float*)d_in[0];
    const float* mask = (const float*)d_in[1];
    const int*   dpos = (const int*)d_in[2];
    const float* Wq = (const float*)d_in[3];  const float* bq = (const float*)d_in[4];
    const float* Wk = (const float*)d_in[5];  const float* bk = (const float*)d_in[6];
    const float* Wv = (const float*)d_in[7];  const float* bv = (const float*)d_in[8];
    const float* W0 = (const float*)d_in[9];  const float* b0 = (const float*)d_in[10];
    const float* W1 = (const float*)d_in[11]; const float* b1 = (const float*)d_in[12];

    char* ws = (char*)d_ws;
    const size_t NX = (size_t)BDIM * SDIM * HDIM;   // 18,874,368
    const size_t NW = (size_t)HDIM * HDIM;          // 589,824
    const size_t NS = (size_t)BDIM * SDIM * SDIM;   // 12,582,912

    size_t off = 0;
    unsigned short* xb  = (unsigned short*)(ws + off); off += NX * 2;
    unsigned short* wqb = (unsigned short*)(ws + off); off += NW * 2;
    unsigned short* wkb = (unsigned short*)(ws + off); off += NW * 2;
    unsigned short* wvb = (unsigned short*)(ws + off); off += NW * 2;
    unsigned short* w0b = (unsigned short*)(ws + off); off += NW * 2;
    unsigned short* w1b = (unsigned short*)(ws + off); off += NW * 2;
    unsigned short* qb  = (unsigned short*)(ws + off); off += NX * 2;
    unsigned short* kb  = (unsigned short*)(ws + off); off += NX * 2;
    unsigned short* vt  = (unsigned short*)(ws + off); off += NX * 2;  // [B][H][S]
    float*          sc  = (float*)(ws + off);          off += NS * 4;
    // Safe aliasing (sequenced dependencies):
    unsigned short* pbuf = qb;                  // P written after Q last read
    unsigned short* resb = (unsigned short*)sc; // res written after scores last read
    unsigned short* hb   = kb;                  // h written after K last read

    const dim3 blk(256);
    // 1) fp32 -> bf16 conversions
    f2bf_kernel<<<dim3(4096), blk, 0, stream>>>(x, xb, (int)NX);
    f2bf_kernel<<<dim3(1024), blk, 0, stream>>>(Wq, wqb, (int)NW);
    f2bf_kernel<<<dim3(1024), blk, 0, stream>>>(Wk, wkb, (int)NW);
    f2bf_kernel<<<dim3(1024), blk, 0, stream>>>(Wv, wvb, (int)NW);
    f2bf_kernel<<<dim3(1024), blk, 0, stream>>>(W0, w0b, (int)NW);
    f2bf_kernel<<<dim3(1024), blk, 0, stream>>>(W1, w1b, (int)NW);

    const long long sX = (long long)SDIM * HDIM;  // per-batch token stride
    const long long sS = (long long)SDIM * SDIM;

    // 2) Q = x Wq^T + bq    (flat M = B*S)
    gemm_bf16_wmma<0, 0, false><<<dim3(96, 6, 1), blk, 0, stream>>>(
        xb, 0, HDIM, wqb, 0, HDIM, qb, 0, HDIM, bq, nullptr, 0, HDIM, 1.0f);
    // 3) K = x Wk^T + bk
    gemm_bf16_wmma<0, 0, false><<<dim3(96, 6, 1), blk, 0, stream>>>(
        xb, 0, HDIM, wkb, 0, HDIM, kb, 0, HDIM, bk, nullptr, 0, HDIM, 1.0f);
    // 4) V^T = (x Wv^T + bv)^T  -> vt[B][H][S]
    gemm_bf16_wmma<0, 0, true><<<dim3(2, 6, BDIM), blk, 0, stream>>>(
        xb, sX, HDIM, wvb, 0, HDIM, vt, sX, SDIM, bv, nullptr, 0, HDIM, 1.0f);
    // 5) scores = (Q K^T) / sqrt(H)   per batch -> f32
    const float scl = 1.0f / sqrtf((float)HDIM);
    gemm_bf16_wmma<1, 0, false><<<dim3(2, 4, BDIM), blk, 0, stream>>>(
        qb, sX, HDIM, kb, sX, HDIM, sc, sS, SDIM, nullptr, nullptr, 0, HDIM, scl);
    // 6) masked softmax -> P (bf16)
    softmax_mask_kernel<<<dim3(BDIM * SDIM / 8), blk, 0, stream>>>(sc, pbuf, mask, dpos);
    // 7) res = P V    (B operand = V^T, K-major)
    gemm_bf16_wmma<0, 0, false><<<dim3(2, 6, BDIM), blk, 0, stream>>>(
        pbuf, sS, SDIM, vt, sX, SDIM, resb, sX, HDIM, nullptr, nullptr, 0, SDIM, 1.0f);
    // 8) h = tanh(res W0^T + b0)
    gemm_bf16_wmma<0, 1, false><<<dim3(96, 6, 1), blk, 0, stream>>>(
        resb, 0, HDIM, w0b, 0, HDIM, hb, 0, HDIM, b0, nullptr, 0, HDIM, 1.0f);
    // 9) out = h W1^T + b1 + x   (f32)
    gemm_bf16_wmma<1, 2, false><<<dim3(96, 6, 1), blk, 0, stream>>>(
        hb, 0, HDIM, w1b, 0, HDIM, d_out, 0, HDIM, b1, x, HDIM, HDIM, 1.0f);
}
